// multihead_attention_89446988906729
// MI455X (gfx1250) — compile-verified, matrix operand order checked by reading, not run on hardware
//
#include <hip/hip_runtime.h>
#include <hip/hip_bf16.h>

// MHA: Q/K/V proj (bf16 WMMA, reg-pipelined) -> flash attention (bf16 WMMA,
// fp32 accum, double-buffered async global->LDS staging, WMMA row-sums)
// -> residual+LN.  B=4, S=1024, D=1024, H=16, DH=64

typedef __bf16 bf16;
typedef __bf16 v16bf __attribute__((ext_vector_type(16)));
typedef float  v8f   __attribute__((ext_vector_type(8)));

#define B_  4
#define S_  1024
#define D_  1024
#define H_  16
#define DH_ 64

// Load a 16(M) x 32(K) bf16 fragment from LDS into the CDNA5 WMMA A/B VGPR
// layout (wave32): lane L<16 -> row m=L, lanes 16..31 -> same rows, K 8..15 and 24..31.
__device__ __forceinline__ v16bf load_frag(const bf16* base, int ldk, int lane) {
    const int m    = lane & 15;
    const int half = lane >> 4;
    const bf16* row = base + m * ldk + half * 8;
    v16bf a;
    ((uint4*)&a)[0] = *(const uint4*)(row);
    ((uint4*)&a)[1] = *(const uint4*)(row + 16);
    return a;
}

__device__ __forceinline__ unsigned pk2(float lo, float hi) {
    union { bf16 h[2]; unsigned u; } c;
    c.h[0] = (bf16)lo; c.h[1] = (bf16)hi;
    return c.u;
}

// gfx1250 async global->LDS copy (16B per lane), tracked by ASYNCcnt.
__device__ __forceinline__ void async_ld_b128(void* lds_dst, const void* gsrc) {
    unsigned ldso = (unsigned)(uintptr_t)lds_dst;   // low 32 bits = LDS offset
    asm volatile("global_load_async_to_lds_b128 %0, %1, off"
                 :: "v"(ldso), "v"(gsrc) : "memory");
}
__device__ __forceinline__ void wait_async0() {
    asm volatile("s_wait_asynccnt 0x0" ::: "memory");
}
// Async loads complete in order: waiting to <=4 outstanding guarantees the
// previous chunk's 4 loads landed while the newest 4 stay in flight.
__device__ __forceinline__ void wait_async4() {
    asm volatile("s_wait_asynccnt 0x4" ::: "memory");
}

// ---------------------------------------------------------------------------
// Kernel 1: Q/K/V projection (templated).  C[4096x1024] = X @ W.
// 128x128 tile / 256 threads, K stepped by 32, register-staged pipeline:
// global loads of tile k+1 overlap WMMAs on tile k.
// WHICH<2 -> head-major [H,B,S,DH]; WHICH==2 -> transposed [H,B,DH,S].
// ---------------------------------------------------------------------------
template <int WHICH>
__global__ __launch_bounds__(256) void qkv_proj_kernel(
    const float* __restrict__ X, const float* __restrict__ W,
    bf16* __restrict__ dst)
{
    __shared__ bf16 At[128 * 32];   // X tile, row-major [m][k]
    __shared__ bf16 Bt[128 * 32];   // W tile transposed: [n][k]

    const int m0   = blockIdx.x * 128;
    const int n0   = blockIdx.y * 128;
    const int t    = threadIdx.x;
    const int lane = t & 31;
    const int w    = t >> 5;
    const int wm   = w >> 2;   // 0..1 : 64 rows each
    const int wn   = w & 3;    // 0..3 : 32 cols each

    // staging geometry
    const int mm  = t >> 1;
    const int ksA = (t & 1) * 16;
    const float* Abase = X + (size_t)(m0 + mm) * D_ + ksA;
    const int n4  = (t & 31) * 4;
    const int kk0 = t >> 5;
    const float* Bbase = W + n0 + n4;

    float4 ra[4], rb[4];
    auto fetch = [&](int k0) {
        const float4* as = (const float4*)(Abase + k0);
        ra[0] = as[0]; ra[1] = as[1]; ra[2] = as[2]; ra[3] = as[3];
        #pragma unroll
        for (int it = 0; it < 4; ++it)
            rb[it] = *(const float4*)(Bbase + (size_t)(k0 + kk0 + it * 8) * D_);
    };
    auto stage = [&]() {
        uint4* adst = (uint4*)(At + mm * 32 + ksA);
        uint4 u0, u1;
        u0.x = pk2(ra[0].x, ra[0].y); u0.y = pk2(ra[0].z, ra[0].w);
        u0.z = pk2(ra[1].x, ra[1].y); u0.w = pk2(ra[1].z, ra[1].w);
        u1.x = pk2(ra[2].x, ra[2].y); u1.y = pk2(ra[2].z, ra[2].w);
        u1.z = pk2(ra[3].x, ra[3].y); u1.w = pk2(ra[3].z, ra[3].w);
        adst[0] = u0; adst[1] = u1;
        #pragma unroll
        for (int it = 0; it < 4; ++it) {
            const int kk = kk0 + it * 8;
            Bt[(n4 + 0) * 32 + kk] = (bf16)rb[it].x;
            Bt[(n4 + 1) * 32 + kk] = (bf16)rb[it].y;
            Bt[(n4 + 2) * 32 + kk] = (bf16)rb[it].z;
            Bt[(n4 + 3) * 32 + kk] = (bf16)rb[it].w;
        }
    };

    const v8f vzero = {};
    v8f acc[4][2];
    #pragma unroll
    for (int i = 0; i < 4; ++i)
        #pragma unroll
        for (int j = 0; j < 2; ++j) acc[i][j] = vzero;

    fetch(0);
    for (int k0 = 0; k0 < D_; k0 += 32) {
        stage();
        __syncthreads();
        if (k0 + 32 < D_) fetch(k0 + 32);   // overlap with WMMAs below

        v16bf a[4], bfr[2];
        #pragma unroll
        for (int i = 0; i < 4; ++i)
            a[i] = load_frag(At + (wm * 64 + i * 16) * 32, 32, lane);
        #pragma unroll
        for (int j = 0; j < 2; ++j)
            bfr[j] = load_frag(Bt + (wn * 32 + j * 16) * 32, 32, lane);

        #pragma unroll
        for (int i = 0; i < 4; ++i)
            #pragma unroll
            for (int j = 0; j < 2; ++j)
                acc[i][j] = __builtin_amdgcn_wmma_f32_16x16x32_bf16(
                    false, a[i], false, bfr[j], (short)0, acc[i][j], false, false);
        __syncthreads();
    }

    // ---- branch-free epilogue (layout chosen at compile time) ----
    const int half = lane >> 4;
    const int nin  = lane & 15;
    #pragma unroll
    for (int i = 0; i < 4; ++i) {
        #pragma unroll
        for (int j = 0; j < 2; ++j) {
            #pragma unroll
            for (int r = 0; r < 8; ++r) {
                const int mg = m0 + wm * 64 + i * 16 + r + 8 * half;
                const int ng = n0 + wn * 32 + j * 16 + nin;
                const int bb = mg >> 10, ss = mg & 1023;   // S_=1024
                const int hh = ng >> 6,  dd = ng & 63;     // DH_=64
                const bf16 v = (bf16)acc[i][j][r];
                if constexpr (WHICH == 2)
                    dst[((size_t)(hh * B_ + bb) * DH_ + dd) * S_ + ss] = v;  // V^T
                else
                    dst[((size_t)(hh * B_ + bb) * S_ + ss) * DH_ + dd] = v;
            }
        }
    }
}

// ---------------------------------------------------------------------------
// Kernel 2: flash attention per (h,b).  Block = 128 threads (4 waves), each
// wave owns a 16-query tile.  K/V^T tiles double-buffered via async DMA so
// chunk i+1 streams in under chunk i's WMMAs.  Row sums computed with a
// ones-matrix WMMA (no shuffle reduction); only the row max uses bpermute.
// ---------------------------------------------------------------------------
__global__ __launch_bounds__(128) void attention_kernel(
    const bf16* __restrict__ qws, const bf16* __restrict__ kws,
    const bf16* __restrict__ vws, const int* __restrict__ query_masks,
    const int* __restrict__ key_masks, float* __restrict__ ows)
{
    __shared__ bf16 Qt[64 * 64];        // 64 queries x 64 dh
    __shared__ bf16 Kt[2][32 * 64];     // 32 keys x 64 dh, double buffered
    __shared__ bf16 Vt[2][64 * 32];     // V^T: 64 dh x 32 keys, double buffered
    __shared__ bf16 Pt[4][16 * 32];     // per-wave P scratch

    const int hb   = blockIdx.y;        // h*B + b
    const int h    = hb >> 2;
    const int b    = hb & 3;
    const int q0   = blockIdx.x * 64;
    const int t    = threadIdx.x;
    const int lane = t & 31;
    const int w    = t >> 5;
    const int half = lane >> 4;
    const int nin  = lane & 15;

    auto issue_chunk = [&](int kc, int buf) {
        const char* ksrc = (const char*)(kws + (size_t)(hb * S_ + kc) * DH_);
        char* kdst = (char*)Kt[buf];
        async_ld_b128(kdst + t * 16,        ksrc + t * 16);
        async_ld_b128(kdst + t * 16 + 2048, ksrc + t * 16 + 2048);
        const int row = t >> 1;
        const int seg = (t & 1) * 16;                 // 16 bf16 = 32 B
        const char* vsrc = (const char*)(vws + (size_t)(hb * DH_ + row) * S_ + kc + seg);
        char* vdst = (char*)(Vt[buf] + row * 32 + seg);
        async_ld_b128(vdst,      vsrc);
        async_ld_b128(vdst + 16, vsrc + 16);
    };

    // ---- stage Q tile + first K/V chunk ----
    {
        const char* src = (const char*)(qws + (size_t)(hb * S_ + q0) * DH_);
        char* dstp = (char*)Qt;
        #pragma unroll
        for (int p = 0; p < 4; ++p)
            async_ld_b128(dstp + (t + p * 128) * 16, src + (t + p * 128) * 16);
    }
    issue_chunk(0, 0);
    wait_async0();
    __syncthreads();

    const v16bf aq0 = load_frag(Qt + (w * 16) * 64,      64, lane);  // dh 0..31
    const v16bf aq1 = load_frag(Qt + (w * 16) * 64 + 32, 64, lane);  // dh 32..63

    v16bf ones;                          // B = all-ones 32x16 for row sums
    #pragma unroll
    for (int i = 0; i < 16; ++i) ones[i] = (bf16)1.0f;

    const v8f vzero = {};
    v8f accO[4];
    #pragma unroll
    for (int u = 0; u < 4; ++u) accO[u] = vzero;
    v8f lacc = vzero;                    // running softmax denominator
    float mrun[8];
    #pragma unroll
    for (int r = 0; r < 8; ++r) mrun[r] = -3.0e38f;

    for (int ic = 0; ic < S_ / 32; ++ic) {
        const int kc  = ic * 32;
        const int buf = ic & 1;
        if (ic + 1 < S_ / 32) {          // stream next chunk under this compute
            issue_chunk(kc + 32, buf ^ 1);
            wait_async4();
        } else {
            wait_async0();
        }
        __syncthreads();

        const int km0 = key_masks[b * S_ + kc + nin];
        const int km1 = key_masks[b * S_ + kc + 16 + nin];

        // ---- scores: 16q x 32k, contraction dh=64 in two K=32 halves ----
        const bf16* Kb = Kt[buf];
        v8f s0 = vzero, s1 = vzero;
        {
            const v16bf b00 = load_frag(Kb,                64, lane);
            const v16bf b01 = load_frag(Kb + 32,           64, lane);
            const v16bf b10 = load_frag(Kb + 16 * 64,      64, lane);
            const v16bf b11 = load_frag(Kb + 16 * 64 + 32, 64, lane);
            s0 = __builtin_amdgcn_wmma_f32_16x16x32_bf16(false, aq0, false, b00, (short)0, s0, false, false);
            s0 = __builtin_amdgcn_wmma_f32_16x16x32_bf16(false, aq1, false, b01, (short)0, s0, false, false);
            s1 = __builtin_amdgcn_wmma_f32_16x16x32_bf16(false, aq0, false, b10, (short)0, s1, false, false);
            s1 = __builtin_amdgcn_wmma_f32_16x16x32_bf16(false, aq1, false, b11, (short)0, s1, false, false);
        }

        // ---- online softmax: max via 16-lane bpermute tree (covers two
        //      rows per register), running rescale folded into accO/lacc ----
        #pragma unroll
        for (int r = 0; r < 8; ++r) {
            float sc0 = km0 ? s0[r] * 0.125f : -1.0e7f;   // scale then key-mask replace
            float sc1 = km1 ? s1[r] * 0.125f : -1.0e7f;
            float hm = fmaxf(sc0, sc1);
            #pragma unroll
            for (int d = 1; d < 16; d <<= 1) hm = fmaxf(hm, __shfl_xor(hm, d, 32));
            const float mnew = fmaxf(mrun[r], hm);
            const float corr = __expf(mrun[r] - mnew);
            const float p0   = __expf(sc0 - mnew);
            const float p1   = __expf(sc1 - mnew);
            mrun[r] = mnew;
            lacc[r] *= corr;
            #pragma unroll
            for (int u = 0; u < 4; ++u) accO[u][r] *= corr;
            const int mrow = r + 8 * half;
            Pt[w][mrow * 32 + nin]      = (bf16)p0;
            Pt[w][mrow * 32 + 16 + nin] = (bf16)p1;
        }

        // ---- O += P (16x32) @ V (32 x 64dh); l += P @ ones ----
        const v16bf ap = load_frag(&Pt[w][0], 32, lane);
        const bf16* Vb = Vt[buf];
        #pragma unroll
        for (int u = 0; u < 4; ++u) {
            const v16bf bv = load_frag(Vb + (u * 16) * 32, 32, lane);
            accO[u] = __builtin_amdgcn_wmma_f32_16x16x32_bf16(
                false, ap, false, bv, (short)0, accO[u], false, false);
        }
        lacc = __builtin_amdgcn_wmma_f32_16x16x32_bf16(
            false, ap, false, ones, (short)0, lacc, false, false);
        __syncthreads();
    }

    // ---- epilogue: 1/l, query mask (post-softmax multiplicative), write O ----
    #pragma unroll
    for (int r = 0; r < 8; ++r) {
        const int mrow = r + 8 * half;
        const int qrow = q0 + w * 16 + mrow;
        const float l   = lacc[r];
        const float inv = (l > 0.f) ? (1.0f / l) : 0.f;
        const float fac = query_masks[b * S_ + qrow] ? inv : 0.f;
        #pragma unroll
        for (int u = 0; u < 4; ++u) {
            ows[((size_t)(b * S_ + qrow)) * D_ + h * DH_ + u * 16 + nin] = accO[u][r] * fac;
        }
    }
}

// ---------------------------------------------------------------------------
// Kernel 3: residual + LayerNorm (eps = 1e-3).  One row (D=1024) per block.
// ---------------------------------------------------------------------------
__global__ __launch_bounds__(256) void resid_ln_kernel(
    const float* __restrict__ queries, const float* __restrict__ ows,
    const float* __restrict__ gamma, const float* __restrict__ beta,
    float* __restrict__ out)
{
    __shared__ float red1[8], red2[8];
    const int row = blockIdx.x;
    const int t   = threadIdx.x;

    const float4 q4 = ((const float4*)(queries + (size_t)row * D_))[t];
    const float4 o4 = ((const float4*)(ows     + (size_t)row * D_))[t];
    float4 r4;
    r4.x = q4.x + o4.x; r4.y = q4.y + o4.y;
    r4.z = q4.z + o4.z; r4.w = q4.w + o4.w;

    float s1 = r4.x + r4.y + r4.z + r4.w;
    float s2 = r4.x * r4.x + r4.y * r4.y + r4.z * r4.z + r4.w * r4.w;
    #pragma unroll
    for (int d = 1; d < 32; d <<= 1) {
        s1 += __shfl_xor(s1, d, 32);
        s2 += __shfl_xor(s2, d, 32);
    }
    if ((t & 31) == 0) { red1[t >> 5] = s1; red2[t >> 5] = s2; }
    __syncthreads();
    float ts1 = 0.f, ts2 = 0.f;
    #pragma unroll
    for (int i = 0; i < 8; ++i) { ts1 += red1[i]; ts2 += red2[i]; }

    const float mean = ts1 * (1.0f / D_);
    const float var  = ts2 * (1.0f / D_) - mean * mean;
    const float rstd = rsqrtf(var + 1e-3f);

    const float4 g4 = ((const float4*)gamma)[t];
    const float4 b4 = ((const float4*)beta)[t];
    float4 o;
    o.x = g4.x * (r4.x - mean) * rstd + b4.x;
    o.y = g4.y * (r4.y - mean) * rstd + b4.y;
    o.z = g4.z * (r4.z - mean) * rstd + b4.z;
    o.w = g4.w * (r4.w - mean) * rstd + b4.w;
    ((float4*)(out + (size_t)row * D_))[t] = o;
}

// ---------------------------------------------------------------------------
extern "C" void kernel_launch(void* const* d_in, const int* in_sizes, int n_in,
                              void* d_out, int out_size, void* d_ws, size_t ws_size,
                              hipStream_t stream) {
    const float* queries = (const float*)d_in[0];
    const float* keys    = (const float*)d_in[1];
    const float* values  = (const float*)d_in[2];
    const int*   qmask   = (const int*)d_in[3];
    const int*   kmask   = (const int*)d_in[4];
    const float* Wq      = (const float*)d_in[5];
    const float* Wk      = (const float*)d_in[6];
    const float* Wv      = (const float*)d_in[7];
    const float* gamma   = (const float*)d_in[8];
    const float* beta    = (const float*)d_in[9];
    float*       out     = (float*)d_out;

    const size_t QKV = (size_t)H_ * B_ * S_ * DH_;   // 4 Mi elements
    bf16*  qws = (bf16*)d_ws;
    bf16*  kws = qws + QKV;
    bf16*  vws = kws + QKV;
    float* ows = (float*)(vws + QKV);

    dim3 g1(B_ * S_ / 128, D_ / 128);
    qkv_proj_kernel<0><<<g1, dim3(256), 0, stream>>>(queries, Wq, qws);
    qkv_proj_kernel<1><<<g1, dim3(256), 0, stream>>>(keys,    Wk, kws);
    qkv_proj_kernel<2><<<g1, dim3(256), 0, stream>>>(values,  Wv, vws);

    dim3 g2(S_ / 64, H_ * B_);
    attention_kernel<<<g2, dim3(128), 0, stream>>>(qws, kws, vws, qmask, kmask, ows);

    resid_ln_kernel<<<dim3(B_ * S_), dim3(256), 0, stream>>>(queries, ows, gamma, beta, out);
}
